// MLPGraphNetwork_30227979829768
// MI455X (gfx1250) — matrix operand order repaired and checked by reading, not compiled, not run on hardware
//
#include <hip/hip_runtime.h>
#include <hip/hip_bf16.h>

// ---------------------------------------------------------------------------
// MLPGraphNetwork on gfx1250 (MI455X): f16 WMMA (f32 accumulate) for the
// edge/node MLPs, LDS-resident weights, atomic scatter for segment_sum.
// ---------------------------------------------------------------------------

typedef __attribute__((ext_vector_type(16))) _Float16 v16h;
typedef __attribute__((ext_vector_type(8)))  float    v8f;

#define N_NODES 50000
#define N_EDGES 800000
#define OUTF    128

// LDS row strides (in halves); must keep rows 16-byte aligned.
#define SP_E 144   // edge kernel activation row stride (max width 128)
#define SP_N 176   // node kernel activation row stride (max width 160)

// Edge-kernel LDS layout (halves)
#define E_W1 0                    // 64*128   = 8192
#define E_W2 (E_W1 + 64 * 128)    // 8192
#define E_W3 (E_W2 + 128 * 128)   // 24576
#define E_W4 (E_W3 + 128 * 128)   // 40960
#define E_ACT (E_W4 + 128 * 128)  // 57344
#define E_LDS_HALVES (E_ACT + 4 * 16 * SP_E)   // 66560 -> 133120 B

// Node-kernel LDS layout (halves)
#define N_W1 0                    // 160*128  = 20480
#define N_W2 (N_W1 + 160 * 128)
#define N_W3 (N_W2 + 128 * 128)
#define N_W4 (N_W3 + 128 * 128)
#define N_ACT (N_W4 + 128 * 128)  // 69632
#define N_LDS_HALVES (N_ACT + 4 * 16 * SP_N)   // 80896 -> 161792 B

// ---------------------------------------------------------------------------
// Weight staging: fp32 [fi,128] row-major  ->  f16 B-fragment order in LDS.
// 16-bit B layout (32x16 per fragment): lane l<16 holds column N=l,
// K=kt*32+{0..7,16..23} packed low->high across 8 VGPRs; lane l>=16 holds
// N=l-16, K=kt*32+{8..15,24..31}.  Fragment f = nt*nKT + kt.
// ---------------------------------------------------------------------------
__device__ __forceinline__ void stage_weights(const float* __restrict__ W,
                                              _Float16* __restrict__ dst,
                                              int fi) {
  const int nKT   = fi >> 5;
  const int total = fi * 128;
  for (int idx = threadIdx.x; idx < total; idx += blockDim.x) {
    int i    = idx & 15;
    int lane = (idx >> 4) & 31;
    int f    = idx >> 9;
    int kt   = f % nKT;
    int nt   = f / nKT;
    int n = nt * 16 + (lane & 15);
    int k = kt * 32 + ((lane >> 4) << 3) + (i & 7) + ((i >> 3) << 4);
    dst[idx] = (_Float16)W[k * 128 + n];
  }
}

// A-fragment (16x32 f16): lane l<16 -> row M=l, K = base+{0..7, 16..23};
// lane l>=16 -> M=l-16, K = base+{8..15, 24..31}.  Two 16B LDS loads.
__device__ __forceinline__ v16h load_a_frag(const _Float16* act, int sp,
                                            int lane, int kbase) {
  const int m    = lane & 15;
  const int koff = (lane >> 4) * 8;
  const _Float16* p = act + m * sp + kbase + koff;
  v16h a;
  ((float4*)&a)[0] = *(const float4*)(p);
  ((float4*)&a)[1] = *(const float4*)(p + 16);
  return a;
}

__device__ __forceinline__ v16h load_b_frag(const _Float16* wlds, int frag,
                                            int lane) {
  const _Float16* p = wlds + ((frag * 32 + lane) << 4);
  v16h b;
  ((float4*)&b)[0] = *(const float4*)(p);
  ((float4*)&b)[1] = *(const float4*)(p + 8);
  return b;
}

// One 16xK @ Kx128 layer tile: 8 f32 accumulators, K = NKT*32.
template <int NKT>
__device__ __forceinline__ void mlp_core(const _Float16* act, int sp,
                                         const _Float16* wlds,
                                         const float* __restrict__ bias,
                                         int lane, v8f acc[8]) {
  const int ncol = lane & 15;
#pragma unroll
  for (int nt = 0; nt < 8; ++nt) {
    float bv = bias[nt * 16 + ncol];
#pragma unroll
    for (int j = 0; j < 8; ++j) acc[nt][j] = bv;
  }
#pragma unroll
  for (int kt = 0; kt < NKT; ++kt) {
    v16h a = load_a_frag(act, sp, lane, kt * 32);
#pragma unroll
    for (int nt = 0; nt < 8; ++nt) {
      v16h b = load_b_frag(wlds, nt * NKT + kt, lane);
      acc[nt] = __builtin_amdgcn_wmma_f32_16x16x32_f16(
          false, a, false, b, (short)0, acc[nt], false, false);
    }
  }
}

// Hidden layer: ReLU + f16 write-back (in place; same-wave DS ops are
// in-order, and all A reads of this layer precede the stores).
template <int NKT>
__device__ __forceinline__ void mlp_layer_relu(_Float16* act, int sp,
                                               const _Float16* wlds,
                                               const float* bias, int lane) {
  v8f acc[8];
  mlp_core<NKT>(act, sp, wlds, bias, lane, acc);
  const int m0 = (lane >> 4) * 8;
  const int ncol = lane & 15;
#pragma unroll
  for (int nt = 0; nt < 8; ++nt)
#pragma unroll
    for (int j = 0; j < 8; ++j) {
      float v = acc[nt][j];
      v = v > 0.f ? v : 0.f;
      act[(m0 + j) * sp + nt * 16 + ncol] = (_Float16)v;
    }
}

// ---------------------------------------------------------------------------
// Edge kernel: e_out = MLP([e, x_src, x_dst, u]); atomic scatter into agg.
// ---------------------------------------------------------------------------
__global__ __launch_bounds__(128) void edge_kernel(
    const float* __restrict__ x, const float* __restrict__ ea,
    const float* __restrict__ u, const int* __restrict__ ei,
    const float* __restrict__ w1, const float* __restrict__ b1,
    const float* __restrict__ w2, const float* __restrict__ b2,
    const float* __restrict__ w3, const float* __restrict__ b3,
    const float* __restrict__ w4, const float* __restrict__ b4,
    float* __restrict__ e_out, float* __restrict__ agg, int E) {
  extern __shared__ _Float16 sm[];
  stage_weights(w1, sm + E_W1, 64);
  stage_weights(w2, sm + E_W2, 128);
  stage_weights(w3, sm + E_W3, 128);
  stage_weights(w4, sm + E_W4, 128);
  __syncthreads();

  const int lane = threadIdx.x & 31;
  const int wid  = threadIdx.x >> 5;
  _Float16* act  = sm + E_ACT + wid * 16 * SP_E;
  const int* src = ei;
  const int* dst = ei + E;

  const int nTiles = E >> 4;
  const int nWaves = (gridDim.x * blockDim.x) >> 5;
  const int gWave  = (blockIdx.x * blockDim.x + threadIdx.x) >> 5;

  for (int t = gWave; t < nTiles; t += nWaves) {
    const int e0 = t << 4;
    // Stage 16x64 f16 input tile: [e(16) | x_src(16) | x_dst(16) | u(16)]
    for (int idx = lane; idx < 16 * 64; idx += 32) {
      int r = idx >> 6, c = idx & 63;
      int e = e0 + r;
      float v;
      if (c < 16)       v = ea[e * 16 + c];
      else if (c < 32)  v = x[src[e] * 16 + (c - 16)];
      else if (c < 48)  v = x[dst[e] * 16 + (c - 32)];
      else              v = u[c - 48];
      act[r * SP_E + c] = (_Float16)v;
    }
    mlp_layer_relu<2>(act, SP_E, sm + E_W1, b1, lane);
    mlp_layer_relu<4>(act, SP_E, sm + E_W2, b2, lane);
    mlp_layer_relu<4>(act, SP_E, sm + E_W3, b3, lane);
    v8f acc[8];
    mlp_core<4>(act, SP_E, sm + E_W4, b4, lane, acc);

    const int m0 = (lane >> 4) * 8;
    const int ncol = lane & 15;
#pragma unroll
    for (int j = 0; j < 8; ++j) {
      const int e = e0 + m0 + j;
      const int d = dst[e];
      float* orow = e_out + (size_t)e * OUTF + ncol;
      float* arow = agg + (size_t)d * OUTF + ncol;
#pragma unroll
      for (int nt = 0; nt < 8; ++nt) {
        orow[nt * 16] = acc[nt][j];
        atomicAdd(arow + nt * 16, acc[nt][j]);
      }
    }
  }
}

// ---------------------------------------------------------------------------
// Node kernel: n_out = MLP([x, agg, u]).  agg aliases n_out storage: each
// tile reads only its own rows before overwriting them -> race free.
// ---------------------------------------------------------------------------
__global__ __launch_bounds__(128) void node_kernel(
    const float* __restrict__ x, const float* __restrict__ u,
    const float* __restrict__ w1, const float* __restrict__ b1,
    const float* __restrict__ w2, const float* __restrict__ b2,
    const float* __restrict__ w3, const float* __restrict__ b3,
    const float* __restrict__ w4, const float* __restrict__ b4,
    float* __restrict__ n_out /* == agg */, int N) {
  extern __shared__ _Float16 sm[];
  stage_weights(w1, sm + N_W1, 160);
  stage_weights(w2, sm + N_W2, 128);
  stage_weights(w3, sm + N_W3, 128);
  stage_weights(w4, sm + N_W4, 128);
  __syncthreads();

  const int lane = threadIdx.x & 31;
  const int wid  = threadIdx.x >> 5;
  _Float16* act  = sm + N_ACT + wid * 16 * SP_N;

  const int nTiles = N >> 4;
  const int nWaves = (gridDim.x * blockDim.x) >> 5;
  const int gWave  = (blockIdx.x * blockDim.x + threadIdx.x) >> 5;

  for (int t = gWave; t < nTiles; t += nWaves) {
    const int n0 = t << 4;
    // Stage 16x160 f16 input tile: [x(16) | agg(128) | u(16)]
    for (int idx = lane; idx < 16 * 160; idx += 32) {
      int r = idx / 160, c = idx - r * 160;
      int nd = n0 + r;
      float v;
      if (c < 16)       v = x[nd * 16 + c];
      else if (c < 144) v = n_out[(size_t)nd * OUTF + (c - 16)];
      else              v = u[c - 144];
      act[r * SP_N + c] = (_Float16)v;
    }
    mlp_layer_relu<5>(act, SP_N, sm + N_W1, b1, lane);
    mlp_layer_relu<4>(act, SP_N, sm + N_W2, b2, lane);
    mlp_layer_relu<4>(act, SP_N, sm + N_W3, b3, lane);
    v8f acc[8];
    mlp_core<4>(act, SP_N, sm + N_W4, b4, lane, acc);

    const int m0 = (lane >> 4) * 8;
    const int ncol = lane & 15;
#pragma unroll
    for (int j = 0; j < 8; ++j) {
      float* orow = n_out + (size_t)(n0 + m0 + j) * OUTF + ncol;
#pragma unroll
      for (int nt = 0; nt < 8; ++nt) orow[nt * 16] = acc[nt][j];
    }
  }
}

// ---------------------------------------------------------------------------
// Column-sum reduction: out128[c] += sum_r m[r*128 + c]
// ---------------------------------------------------------------------------
__global__ __launch_bounds__(128) void colsum_kernel(
    const float* __restrict__ m, int rows, float* __restrict__ out128) {
  const int c = threadIdx.x;  // 128 threads
  float s = 0.f;
  for (int r = blockIdx.x; r < rows; r += gridDim.x)
    s += m[(size_t)r * OUTF + c];
  atomicAdd(&out128[c], s);
}

// ---------------------------------------------------------------------------
// Global MLP: 1 row, exact fp32.  g_in = [u(16), n_sum(128), e_sum(128)]
// ---------------------------------------------------------------------------
__global__ __launch_bounds__(128) void global_kernel(
    const float* __restrict__ u, const float* __restrict__ esum,
    const float* __restrict__ nsum,
    const float* __restrict__ w1, const float* __restrict__ b1,
    const float* __restrict__ w2, const float* __restrict__ b2,
    const float* __restrict__ w3, const float* __restrict__ b3,
    const float* __restrict__ w4, const float* __restrict__ b4,
    float* __restrict__ gout) {
  __shared__ float a0[272];
  __shared__ float a1[128];
  const int t = threadIdx.x;  // 128 threads
  for (int i = t; i < 272; i += 128) {
    float v = (i < 16) ? u[i] : (i < 144) ? nsum[i - 16] : esum[i - 144];
    a0[i] = v;
  }
  __syncthreads();
  float h = b1[t];
  for (int k = 0; k < 272; ++k) h += a0[k] * w1[k * 128 + t];
  h = fmaxf(h, 0.f);
  a1[t] = h;
  __syncthreads();
  h = b2[t];
  for (int k = 0; k < 128; ++k) h += a1[k] * w2[k * 128 + t];
  h = fmaxf(h, 0.f);
  __syncthreads();
  a1[t] = h;
  __syncthreads();
  h = b3[t];
  for (int k = 0; k < 128; ++k) h += a1[k] * w3[k * 128 + t];
  h = fmaxf(h, 0.f);
  __syncthreads();
  a1[t] = h;
  __syncthreads();
  h = b4[t];
  for (int k = 0; k < 128; ++k) h += a1[k] * w4[k * 128 + t];
  gout[t] = h;
}

__global__ void zero_kernel(float* __restrict__ agg, float* __restrict__ sums,
                            int aggN) {
  int i = blockIdx.x * blockDim.x + threadIdx.x;
  if (i < aggN) agg[i] = 0.f;
  if (i < 256) sums[i] = 0.f;
}

extern "C" void kernel_launch(void* const* d_in, const int* in_sizes, int n_in,
                              void* d_out, int out_size, void* d_ws,
                              size_t ws_size, hipStream_t stream) {
  const float* x  = (const float*)d_in[0];
  const float* ea = (const float*)d_in[1];
  const float* u  = (const float*)d_in[2];
  const int*   ei = (const int*)d_in[3];
  const float* ew1 = (const float*)d_in[4];  const float* eb1 = (const float*)d_in[5];
  const float* ew2 = (const float*)d_in[6];  const float* eb2 = (const float*)d_in[7];
  const float* ew3 = (const float*)d_in[8];  const float* eb3 = (const float*)d_in[9];
  const float* ew4 = (const float*)d_in[10]; const float* eb4 = (const float*)d_in[11];
  const float* nw1 = (const float*)d_in[12]; const float* nb1 = (const float*)d_in[13];
  const float* nw2 = (const float*)d_in[14]; const float* nb2 = (const float*)d_in[15];
  const float* nw3 = (const float*)d_in[16]; const float* nb3 = (const float*)d_in[17];
  const float* nw4 = (const float*)d_in[18]; const float* nb4 = (const float*)d_in[19];
  const float* gw1 = (const float*)d_in[20]; const float* gb1 = (const float*)d_in[21];
  const float* gw2 = (const float*)d_in[22]; const float* gb2 = (const float*)d_in[23];
  const float* gw3 = (const float*)d_in[24]; const float* gb3 = (const float*)d_in[25];
  const float* gw4 = (const float*)d_in[26]; const float* gb4 = (const float*)d_in[27];

  const int E = in_sizes[1] / 16;  // 800000
  const int N = in_sizes[0] / 16;  // 50000

  float* e_out = (float*)d_out;
  float* n_out = e_out + (size_t)E * OUTF;  // doubles as agg scratch
  float* g_out = n_out + (size_t)N * OUTF;
  float* e_sum = (float*)d_ws;       // 128 floats
  float* n_sum = e_sum + 128;        // 128 floats

  const int aggN = N * OUTF;
  zero_kernel<<<(aggN + 255) / 256, 256, 0, stream>>>(n_out, e_sum, aggN);

  edge_kernel<<<512, 128, E_LDS_HALVES * sizeof(_Float16), stream>>>(
      x, ea, u, ei, ew1, eb1, ew2, eb2, ew3, eb3, ew4, eb4, e_out, n_out, E);

  colsum_kernel<<<512, 128, 0, stream>>>(e_out, E, e_sum);

  node_kernel<<<512, 128, N_LDS_HALVES * sizeof(_Float16), stream>>>(
      x, u, nw1, nb1, nw2, nb2, nw3, nb3, nw4, nb4, n_out, N);

  colsum_kernel<<<512, 128, 0, stream>>>(n_out, N, n_sum);

  global_kernel<<<1, 128, 0, stream>>>(u, e_sum, n_sum, gw1, gb1, gw2, gb2,
                                       gw3, gb3, gw4, gb4, g_out);
}
